// CausalSelfAttention_28664611734018
// MI455X (gfx1250) — compile-verified, matrix operand order checked by reading, not run on hardware
//
#include <hip/hip_runtime.h>
#include <hip/hip_bf16.h>

typedef __attribute__((ext_vector_type(16))) _Float16 v16h;
typedef __attribute__((ext_vector_type(8)))  _Float16 v8h;
typedef __attribute__((ext_vector_type(4)))  _Float16 v4h;
typedef __attribute__((ext_vector_type(8)))  float    v8f;
typedef __attribute__((ext_vector_type(4)))  float    v4f;

#define T_SEQ   128
#define C_EMB   64
#define N_HEADS 4
#define HEAD_D  16
#define XS      72    // row stride (halfs) for width-64 LDS buffers (144B = 9*16)
#define VS      136   // row stride (halfs) for width-128 LDS buffers (272B = 17*16)

// ---- LDS layout (bytes), lifetime-aliased ----
// [0      .. 34816)  WT_attn [192][XS]  (phase0/1)  | attb [8 waves][16][VS] (phase2)
// [34816  .. 53248)  xh [128][XS] f16   (phase0/1)  | yh  [128][XS]          (phase2/3)
// [53248  .. 90112)  qh/kh [128][XS] each (phase1+) | xstage fp32 32KB       (phase0 only)
// [90112  ..107520)  vT [64][VS]
// [107520 ..116736)  WT_proj [64][XS]
// [116736 ..117504)  b_attn f32[192]
// [117504 ..117760)  b_proj f32[64]
#define SMEM_BYTES 117760

__device__ __forceinline__ v8f wmma16(v16h a, v16h b, v8f c) {
  // D = A(16x32 f16) * B(32x16 f16) + C(16x16 f32)
  return __builtin_amdgcn_wmma_f32_16x16x32_f16(false, a, false, b, (short)0, c, false, false);
}

// Fragment loader for both A (row-major rows=M) and B (row-major B^T, rows=N).
// ISA 16-bit operand layout: lane<16 holds K {kb..kb+7, kb+16..kb+23} with kb=0,
// lane>=16 the same with kb=8.  Two aligned 16B LDS reads per lane.
__device__ __forceinline__ v16h load_frag(const _Float16* base, int stride, int row_off, int k_off) {
  const int lane = threadIdx.x & 31;
  const int m    = lane & 15;
  const int kb   = (lane >> 4) << 3;           // 0 or 8
  const _Float16* p = base + (row_off + m) * stride + k_off + kb;
  const v8h lo = *(const v8h*)__builtin_assume_aligned(p, 16);
  const v8h hi = *(const v8h*)__builtin_assume_aligned(p + 16, 16);
  v16h r;
#pragma unroll
  for (int i = 0; i < 8; ++i) { r[i] = lo[i]; r[i + 8] = hi[i]; }
  return r;
}

// K=16 variant (head_dim=16 padded to K=32 with zeros in the K>=16 halves).
__device__ __forceinline__ v16h load_frag_k16(const _Float16* base, int stride, int row_off, int col_off) {
  const int lane = threadIdx.x & 31;
  const int m    = lane & 15;
  const int kb   = (lane >> 4) << 3;
  const _Float16* p = base + (row_off + m) * stride + col_off + kb;
  const v8h lo = *(const v8h*)__builtin_assume_aligned(p, 16);
  v16h r;
#pragma unroll
  for (int i = 0; i < 8; ++i) { r[i] = lo[i]; r[i + 8] = (_Float16)0.0f; }
  return r;
}

__global__ __launch_bounds__(256)
void CausalSelfAttention_fused_wmma(const float* __restrict__ xg,
                                    const float* __restrict__ Wag,
                                    const float* __restrict__ bag,
                                    const float* __restrict__ Wpg,
                                    const float* __restrict__ bpg,
                                    float* __restrict__ outg) {
  extern __shared__ char smem[];
  _Float16* WTa    = (_Float16*)(smem);            // [192][XS], WTa[n][k] = Wa[k][n]
  _Float16* attb   = (_Float16*)(smem);            // aliases WTa in phase 2
  _Float16* xh     = (_Float16*)(smem + 34816);    // [128][XS]
  _Float16* yh     = (_Float16*)(smem + 34816);    // aliases xh in phase 2/3
  _Float16* qh     = (_Float16*)(smem + 53248);    // [128][XS]
  _Float16* kh     = (_Float16*)(smem + 71680);    // [128][XS]
  float*    xstage = (float*)(smem + 53248);       // fp32 x staging, aliases qh/kh, phase0 only
  _Float16* vT     = (_Float16*)(smem + 90112);    // [64][VS], vT[d][s]
  _Float16* WTp    = (_Float16*)(smem + 107520);   // [64][XS], WTp[n][k] = Wp[k][n]
  float*    ba     = (float*)(smem + 116736);
  float*    bp     = (float*)(smem + 117504);

  const int tid  = threadIdx.x;
  const int b    = blockIdx.x;
  const int w    = tid >> 5;                     // wave id 0..7 -> M-tile
  const int lane = tid & 31;
  const int m    = lane & 15;                    // N / column index inside tile
  const int off8 = (lane >> 4) << 3;             // accumulator row offset: 0 or 8

  const float* xb = xg + (size_t)b * T_SEQ * C_EMB;
  __builtin_prefetch(Wag, 0, 3);

  // ---------------- Phase 0a: async-copy x (fp32, 32KB) into LDS staging ----------------
  {
    const unsigned ldsbase = (unsigned)(uintptr_t)xstage;
#pragma unroll
    for (int it = 0; it < (T_SEQ * C_EMB / 4) / 256; ++it) {        // 8 iterations
      const int i4 = tid + it * 256;
      const unsigned lds_addr = ldsbase + (unsigned)i4 * 16u;
      const unsigned goff     = (unsigned)i4 * 16u;
      asm volatile("global_load_async_to_lds_b128 %0, %1, %2"
                   :: "v"(lds_addr), "v"(goff), "s"(xb) : "memory");
    }
  }

  // ---------------- Phase 0b: stage weights (f16, transposed) while async runs ----------
  for (int i = tid; i < (64 * 192) / 4; i += 256) {                 // 12 iterations
    const v4f fw = *(const v4f*)__builtin_assume_aligned(Wag + i * 4, 16);
    const int k  = (i * 4) / 192;
    const int n0 = (i * 4) - k * 192;
#pragma unroll
    for (int u = 0; u < 4; ++u) WTa[(n0 + u) * XS + k] = (_Float16)fw[u];
  }
  for (int i = tid; i < (64 * 64) / 4; i += 256) {                  // 4 iterations
    const v4f fw = *(const v4f*)__builtin_assume_aligned(Wpg + i * 4, 16);
    const int k  = i >> 4;
    const int n0 = (i & 15) << 2;
#pragma unroll
    for (int u = 0; u < 4; ++u) WTp[(n0 + u) * XS + k] = (_Float16)fw[u];
  }
  if (tid < 192) ba[tid] = bag[tid];
  if (tid < 64)  bp[tid] = bpg[tid];

  // ---------------- Phase 0c: convert own staged float4s to f16 ----------------
  asm volatile("s_wait_asynccnt 0x0" ::: "memory");                 // own async loads done
#pragma unroll
  for (int it = 0; it < (T_SEQ * C_EMB / 4) / 256; ++it) {
    const int i4 = tid + it * 256;
    const v4f f = *(const v4f*)__builtin_assume_aligned(xstage + i4 * 4, 16);
    v4h hc;
#pragma unroll
    for (int u = 0; u < 4; ++u) hc[u] = (_Float16)f[u];
    const int row  = i4 >> 4;
    const int col4 = (i4 & 15) << 2;
    *(v4h*)__builtin_assume_aligned(xh + row * XS + col4, 8) = hc;
  }
  __syncthreads();

  // ---------------- Phase 1: qkv = x @ W_attn + b_attn ----------------
  {
    const v16h a0 = load_frag(xh, XS, w * 16, 0);
    const v16h a1 = load_frag(xh, XS, w * 16, 32);
#pragma unroll
    for (int n = 0; n < 12; ++n) {
      v8f acc = {0.f, 0.f, 0.f, 0.f, 0.f, 0.f, 0.f, 0.f};
      const v16h b0 = load_frag(WTa + n * 16 * XS, XS, 0, 0);
      const v16h b1 = load_frag(WTa + n * 16 * XS, XS, 0, 32);
      acc = wmma16(a0, b0, acc);
      acc = wmma16(a1, b1, acc);
      const int   col  = n * 16 + m;
      const float bias = ba[col];
      if (n < 8) {
        // q / k tiles: row-major scatter (t varies per accumulator row)
        _Float16* dst = (n < 4) ? (qh + col) : (kh + (col - 64));
#pragma unroll
        for (int r = 0; r < 8; ++r) {
          const int t = w * 16 + off8 + r;
          dst[t * XS] = (_Float16)(acc[r] + bias);
        }
      } else {
        // v tile: transposed store; lane's 8 rows are 8 consecutive t -> one b128 store
        v8h hv;
#pragma unroll
        for (int r = 0; r < 8; ++r) hv[r] = (_Float16)(acc[r] + bias);
        *(v8h*)__builtin_assume_aligned(vT + (col - 128) * VS + w * 16 + off8, 16) = hv;
      }
    }
  }
  __syncthreads();

  // ---------------- Phase 2: per-head attention ----------------
  _Float16* attw = attb + w * 16 * VS;           // wave-private [16][VS] prob tile
#pragma unroll 1
  for (int h = 0; h < N_HEADS; ++h) {
    // scores S = (q @ k^T) * 0.25 for row-tile w, all 8 column tiles
    v8f s[8];
    const v16h aq = load_frag_k16(qh, XS, w * 16, h * 16);
#pragma unroll
    for (int j = 0; j < 8; ++j) {
      const v16h bk = load_frag_k16(kh, XS, j * 16, h * 16);
      v8f z = {0.f, 0.f, 0.f, 0.f, 0.f, 0.f, 0.f, 0.f};
      s[j] = wmma16(aq, bk, z);
    }
    // causal mask + softmax; each row lives in one 16-lane half-wave
#pragma unroll
    for (int r = 0; r < 8; ++r) {
      const int t = w * 16 + off8 + r;
      float vals[8];
      float mx = -3.0e38f;
#pragma unroll
      for (int j = 0; j < 8; ++j) {
        const int scol = j * 16 + m;
        float v = s[j][r] * 0.25f;               // 1/sqrt(HEAD_D)
        v = (scol <= t) ? v : -3.0e38f;
        vals[j] = v;
        mx = fmaxf(mx, v);
      }
#pragma unroll
      for (int d = 1; d < 16; d <<= 1) mx = fmaxf(mx, __shfl_xor(mx, d, 32));
      float sum = 0.f;
#pragma unroll
      for (int j = 0; j < 8; ++j) { const float e = __expf(vals[j] - mx); vals[j] = e; sum += e; }
#pragma unroll
      for (int d = 1; d < 16; d <<= 1) sum += __shfl_xor(sum, d, 32);
      const float inv = 1.0f / sum;
#pragma unroll
      for (int j = 0; j < 8; ++j)
        attw[(off8 + r) * VS + j * 16 + m] = (_Float16)(vals[j] * inv);
    }
    // y_tile = att @ v  (K = 128 in 4 steps), via vT as B^T
    v8f acc = {0.f, 0.f, 0.f, 0.f, 0.f, 0.f, 0.f, 0.f};
#pragma unroll
    for (int ks = 0; ks < 4; ++ks) {
      const v16h aa = load_frag(attw, VS, 0, ks * 32);
      const v16h bv = load_frag(vT + h * 16 * VS, VS, 0, ks * 32);
      acc = wmma16(aa, bv, acc);
    }
#pragma unroll
    for (int r = 0; r < 8; ++r) {
      const int t = w * 16 + off8 + r;
      yh[t * XS + h * 16 + m] = (_Float16)acc[r];
    }
  }
  __syncthreads();

  // ---------------- Phase 3: out = y @ W_proj + b_proj ----------------
  {
    float* outp = outg + (size_t)b * T_SEQ * C_EMB;
    const v16h a0 = load_frag(yh, XS, w * 16, 0);
    const v16h a1 = load_frag(yh, XS, w * 16, 32);
#pragma unroll
    for (int n = 0; n < 4; ++n) {
      v8f acc = {0.f, 0.f, 0.f, 0.f, 0.f, 0.f, 0.f, 0.f};
      const v16h b0 = load_frag(WTp + n * 16 * XS, XS, 0, 0);
      const v16h b1 = load_frag(WTp + n * 16 * XS, XS, 0, 32);
      acc = wmma16(a0, b0, acc);
      acc = wmma16(a1, b1, acc);
      const int   col  = n * 16 + m;
      const float bias = bp[col];
#pragma unroll
      for (int r = 0; r < 8; ++r) {
        const int t = w * 16 + off8 + r;
        outp[t * C_EMB + col] = acc[r] + bias;
      }
    }
  }
}

extern "C" void kernel_launch(void* const* d_in, const int* in_sizes, int n_in,
                              void* d_out, int out_size, void* d_ws, size_t ws_size,
                              hipStream_t stream) {
  const float* x  = (const float*)d_in[0];
  const float* Wa = (const float*)d_in[1];
  const float* ba = (const float*)d_in[2];
  const float* Wp = (const float*)d_in[3];
  const float* bp = (const float*)d_in[4];
  float* out = (float*)d_out;

  const int B = in_sizes[0] / (T_SEQ * C_EMB);   // 1024

  hipFuncSetAttribute((const void*)CausalSelfAttention_fused_wmma,
                      hipFuncAttributeMaxDynamicSharedMemorySize, SMEM_BYTES);

  CausalSelfAttention_fused_wmma<<<dim3(B), dim3(256), SMEM_BYTES, stream>>>(
      x, Wa, ba, Wp, bp, out);
}
